// GraphAutoencoder_48773648613743
// MI455X (gfx1250) — matrix operand order repaired and checked by reading, not compiled, not run on hardware
//
#include <hip/hip_runtime.h>

// ---------------------------------------------------------------------------
// 2-layer GCN: gcn_norm + 2x (f16-WMMA GEMM with fused bias/self-loop epilogue
// -> edge gather*norm -> scatter-add via f32 global atomics), ReLU fused into
// the layer-2 GEMM's A-tile load.  The workload is HBM-bound (~10 GB of edge
// gather/scatter traffic), so kernels use 128-bit memory ops throughout and
// the GEMM epilogue writes both h (for gathers) and the self-loop-initialized
// output (saving a full 205 MB/layer init pass).
// ---------------------------------------------------------------------------

typedef __attribute__((ext_vector_type(16))) _Float16 v16h;
typedef __attribute__((ext_vector_type(8)))  _Float16 v8h;
typedef __attribute__((ext_vector_type(8)))  float    v8f;

#define BM 32
#define BN 64
#define BK 64      // two 16x16x32 WMMAs per staging round
#define LDA_S 72   // halves; 144B row stride, 16B aligned, conflict-reducing pad
#define LDB_S 72

// hbuf[M,N] = act(A[M,K]) @ W[K,N]           (act = relu if relu_a)
// out [M,N] = bias[N] + dinv[M]^2 * hbuf[M,N] (self-loop term pre-folded)
// Requires M%32==0, K%64==0, N%64==0 (true: M=100000, K in {1024,256},
// N in {256,128}).
__global__ __launch_bounds__(256) void gemm_f16_wmma(
    const float* __restrict__ A, const float* __restrict__ W,
    const float* __restrict__ dinv, const float* __restrict__ bias,
    float* __restrict__ hbuf, float* __restrict__ out,
    int M, int K, int N, int relu_a)
{
    __shared__ _Float16 sA[BM * LDA_S];
    __shared__ _Float16 sB[BN * LDB_S];

    const int tid   = threadIdx.x;
    const int lane  = tid & 31;
    const int wave  = tid >> 5;      // 0..7
    const int waveM = wave & 1;      // 2 sub-tiles vertically
    const int waveN = wave >> 1;     // 4 sub-tiles horizontally
    const int rowBase = blockIdx.x * BM;
    const int colBase = blockIdx.y * BN;

    const int h   = lane >> 4;       // half-wave id (0/1)
    const int l16 = lane & 15;

    v8f acc = {};

    for (int k0 = 0; k0 < K; k0 += BK) {
        // ---- stage A tile (BM x BK fp32 -> f16, optional fused relu) ----
        #pragma unroll
        for (int t = 0; t < 2; ++t) {
            int idx = (tid + t * 256) * 4;     // 2048 floats total
            int r   = idx >> 6;
            int c   = idx & 63;
            float4 v = *(const float4*)(A + (size_t)(rowBase + r) * K + k0 + c);
            if (relu_a) {
                v.x = fmaxf(v.x, 0.f); v.y = fmaxf(v.y, 0.f);
                v.z = fmaxf(v.z, 0.f); v.w = fmaxf(v.w, 0.f);
            }
            _Float16* p = &sA[r * LDA_S + c];
            p[0] = (_Float16)v.x; p[1] = (_Float16)v.y;
            p[2] = (_Float16)v.z; p[3] = (_Float16)v.w;
        }
        // ---- stage B tile (BK x BN fp32 -> f16, transposed: sB[n][k]) ----
        #pragma unroll
        for (int t = 0; t < 4; ++t) {
            int idx = (tid + t * 256) * 4;     // 4096 floats total
            int kk  = idx >> 6;
            int n   = idx & 63;
            float4 v = *(const float4*)(W + (size_t)(k0 + kk) * N + colBase + n);
            sB[(n + 0) * LDB_S + kk] = (_Float16)v.x;
            sB[(n + 1) * LDB_S + kk] = (_Float16)v.y;
            sB[(n + 2) * LDB_S + kk] = (_Float16)v.z;
            sB[(n + 3) * LDB_S + kk] = (_Float16)v.w;
        }
        __syncthreads();

        // ---- two WMMAs per staging round (K sub-tiles s=0,1) ----
        const int mrow = waveM * 16 + l16;
        const int ncol = waveN * 16 + l16;
        #pragma unroll
        for (int s = 0; s < 2; ++s) {
            // A 16x32: lanes 0-15 hold K {0..7,16..23}, lanes 16-31 K {8..15,24..31}
            int kb0 = s * 32 + h * 8;
            int kb1 = s * 32 + 16 + h * 8;
            v8h alo = *(const v8h*)(&sA[mrow * LDA_S + kb0]);
            v8h ahi = *(const v8h*)(&sA[mrow * LDA_S + kb1]);
            v16h a;
            #pragma unroll
            for (int i = 0; i < 8; ++i) { a[i] = alo[i]; a[8 + i] = ahi[i]; }

            // B 32x16: lanes 0-15 hold K {0..7,16..23} of col l16, lanes 16-31 K {8..15,24..31}
            int kb = s * 32 + h * 8;
            v8h blo = *(const v8h*)(&sB[ncol * LDB_S + kb]);
            v8h bhi = *(const v8h*)(&sB[ncol * LDB_S + kb + 16]);
            v16h b;
            #pragma unroll
            for (int i = 0; i < 8; ++i) { b[i] = blo[i]; b[8 + i] = bhi[i]; }

            acc = __builtin_amdgcn_wmma_f32_16x16x32_f16(
                /*neg_a=*/false, a, /*neg_b=*/false, b,
                /*c_mod=*/(short)0, acc, /*reuse_a=*/false, /*reuse_b=*/false);
        }
        __syncthreads();
    }

    // ---- fused epilogue: C/D layout lane l16 = N col, vgpr r -> M = r+8*h ----
    int col  = colBase + waveN * 16 + l16;
    int row0 = rowBase + waveM * 16 + 8 * h;
    float bcol = bias[col];
    #pragma unroll
    for (int r = 0; r < 8; ++r) {
        size_t off = (size_t)(row0 + r) * N + col;
        float hv = acc[r];
        float s  = dinv[row0 + r];
        hbuf[off] = hv;                 // for edge gathers
        out[off]  = bcol + s * s * hv;  // bias + self-loop contribution
    }
}

// ---------------------------------------------------------------------------
__global__ void deg_init(float* __restrict__ deg, int n)
{
    int i = blockIdx.x * blockDim.x + threadIdx.x;
    if (i < n) deg[i] = 1.0f;   // self-loop weight
}

__global__ void deg_accum(const long long* __restrict__ ei,
                          const float* __restrict__ w,
                          float* __restrict__ deg, int E)
{
    int e = blockIdx.x * blockDim.x + threadIdx.x;
    if (e < E) {
        long long dst = ei[(size_t)E + e];
        __hip_atomic_fetch_add(&deg[dst], w[e],
                               __ATOMIC_RELAXED, __HIP_MEMORY_SCOPE_AGENT);
    }
}

__global__ void dinv_kernel(const float* __restrict__ deg,
                            float* __restrict__ dinv, int n)
{
    int i = blockIdx.x * blockDim.x + threadIdx.x;
    if (i < n) {
        float d = deg[i];
        dinv[i] = d > 0.f ? rsqrtf(d) : 0.f;
    }
}

// one wave per edge: gather h[src] (128-bit), scatter-add norm*h to out[dst]
__global__ __launch_bounds__(256) void agg_edges(
    const long long* __restrict__ ei, const float* __restrict__ w,
    const float* __restrict__ dinv, const float* __restrict__ hbuf,
    float* __restrict__ out, int E, int N)
{
    int gid  = blockIdx.x * blockDim.x + threadIdx.x;
    int wid  = gid >> 5;
    int lane = gid & 31;
    if (wid >= E) return;
    long long src = ei[wid];
    long long dst = ei[(size_t)E + wid];
    float norm = dinv[src] * w[wid] * dinv[dst];
    const float* hs = hbuf + (size_t)src * N;
    float*       od = out  + (size_t)dst * N;
    for (int j = lane * 4; j < N; j += 128) {
        float4 v = *(const float4*)(hs + j);
        __hip_atomic_fetch_add(&od[j + 0], norm * v.x, __ATOMIC_RELAXED, __HIP_MEMORY_SCOPE_AGENT);
        __hip_atomic_fetch_add(&od[j + 1], norm * v.y, __ATOMIC_RELAXED, __HIP_MEMORY_SCOPE_AGENT);
        __hip_atomic_fetch_add(&od[j + 2], norm * v.z, __ATOMIC_RELAXED, __HIP_MEMORY_SCOPE_AGENT);
        __hip_atomic_fetch_add(&od[j + 3], norm * v.w, __ATOMIC_RELAXED, __HIP_MEMORY_SCOPE_AGENT);
    }
}

// ---------------------------------------------------------------------------
extern "C" void kernel_launch(void* const* d_in, const int* in_sizes, int n_in,
                              void* d_out, int out_size, void* d_ws, size_t ws_size,
                              hipStream_t stream)
{
    const float*     x  = (const float*)d_in[0];       // [100000,1024]
    const long long* ei = (const long long*)d_in[1];   // [2,E] int64
    const float*     ew = (const float*)d_in[2];       // [E]
    const float*     W1 = (const float*)d_in[3];       // [1024,256]
    const float*     b1 = (const float*)d_in[4];       // [256]
    const float*     W2 = (const float*)d_in[5];       // [256,128]
    const float*     b2 = (const float*)d_in[6];       // [128]
    float*           z  = (float*)d_out;               // [100000,128]

    const int NODES = 100000;
    const int E     = in_sizes[2];                     // 3,200,000
    const int DIN   = 1024, DHID = 256, DEMB = 128;

    // workspace layout:
    //   [0, 1MB) deg   [1MB, 2MB) dinv
    //   [2MB, 2MB+102.4MB)  h-buffer (h1, reused as h2)
    //   [2MB+102.4MB, ...)  aggregation buffer o1 (102.4MB)
    char*  ws   = (char*)d_ws;
    float* deg  = (float*)(ws);
    float* dinv = (float*)(ws + (1u << 20));
    float* h1   = (float*)(ws + (2u << 20));
    float* o1   = (float*)(ws + (2u << 20) + (size_t)NODES * DHID * 4);
    float* h2   = h1;   // 51.2MB fits in h1's 102.4MB slot

    // ---- gcn_norm: deg = 1 + sum_w(dst), dinv = rsqrt(deg) ----
    deg_init   <<<(NODES + 255) / 256, 256, 0, stream>>>(deg, NODES);
    deg_accum  <<<(E + 255) / 256,     256, 0, stream>>>(ei, ew, deg, E);
    dinv_kernel<<<(NODES + 255) / 256, 256, 0, stream>>>(deg, dinv, NODES);

    // ---- layer 1: h1 = x@W1, o1 = b1 + dinv^2*h1, then edge scatter ----
    dim3 g1(NODES / BM, DHID / BN);
    gemm_f16_wmma<<<g1, 256, 0, stream>>>(x, W1, dinv, b1, h1, o1,
                                          NODES, DIN, DHID, 0);
    agg_edges<<<(int)(((size_t)E * 32 + 255) / 256), 256, 0, stream>>>(
        ei, ew, dinv, h1, o1, E, DHID);

    // ---- layer 2: relu fused into GEMM A-load; epilogue writes z ----
    dim3 g2(NODES / BM, DEMB / BN);
    gemm_f16_wmma<<<g2, 256, 0, stream>>>(o1, W2, dinv, b2, h2, z,
                                          NODES, DHID, DEMB, 1);
    agg_edges<<<(int)(((size_t)E * 32 + 255) / 256), 256, 0, stream>>>(
        ei, ew, dinv, h2, z, E, DEMB);
}
